// GRADATE_51616916963816
// MI455X (gfx1250) — compile-verified, matrix operand order checked by reading, not run on hardware
//
#include <hip/hip_runtime.h>
#include <hip/hip_bf16.h>

// Shapes from the reference
#define BATCH 65536
#define SS    5
#define FF    512
#define HH    64

typedef __attribute__((ext_vector_type(16))) __bf16 v16bf;
typedef __attribute__((ext_vector_type(8)))  __bf16 v8bf;
typedef __attribute__((ext_vector_type(4)))  __bf16 v4bf;
typedef __attribute__((ext_vector_type(8)))  float  v8f;

// ---- Kernel 1 LDS geometry (bank-conflict padded, all 16B aligned) ----
#define LAS 136              // A row stride in bf16 (128 data + 8 pad) -> 272B = 16*17
#define LA_ROWS 80
#define LWS 136              // W row stride in bf16
#define LW_OFF (LA_ROWS*LAS)           // W starts after A (bf16 units)
#define LFS 68               // fts row stride in f32
// union region: max( A+W = (80+128)*136*2 = 56576 B , fts = 2*80*68*4 = 43520 B )
#define SM_BYTES 57344

__device__ __forceinline__ v16bf mk16(const __bf16* p0, const __bf16* p1) {
    const v8bf a = *(const v8bf*)p0;
    const v8bf b = *(const v8bf*)p1;
    v16bf r;
#pragma unroll
    for (int i = 0; i < 8; ++i) { r[i] = a[i]; r[8 + i] = b[i]; }
    return r;
}

// ============================ Kernel 1 ============================
// Per WG: 16 batches (80 rows of bf). fts1/fts2 via bf16 WMMA, then the
// tiny ba(5x5) mix + bias + PReLU + reductions.
__global__ void __launch_bounds__(256)
gcn_wmma_kernel(const float* __restrict__ bf, const float* __restrict__ ba,
                const float* __restrict__ Wc, const float* __restrict__ bc,
                const float* __restrict__ ac,
                const float* __restrict__ Wp, const float* __restrict__ bp,
                const float* __restrict__ ap,
                float* __restrict__ out, float* __restrict__ ws)
{
    __shared__ __align__(16) unsigned char smem[SM_BYTES];
    __shared__ float lBa[16 * 25];

    __bf16* lA = (__bf16*)smem;
    __bf16* lW = lA + LW_OFF;
    float*  lF = (float*)smem;          // reused after WMMA phase

    const int tid = threadIdx.x;
    const int wg  = blockIdx.x;
    const int b0  = wg * 16;
    const size_t rowBase = (size_t)wg * 80;   // row index into (B*S, F)

    if (tid < 400) lBa[tid] = ba[(size_t)b0 * 25 + tid];

    const int wv   = tid >> 5;
    const int lane = tid & 31;
    const int l    = lane & 15;
    const int hi   = lane >> 4;

    v8f acc[5];
#pragma unroll
    for (int i = 0; i < 5; ++i)
#pragma unroll
        for (int j = 0; j < 8; ++j) acc[i][j] = 0.0f;

    for (int kc = 0; kc < 4; ++kc) {          // K chunks of 128
        __syncthreads();
        // A chunk: 80 rows x 128 f32 -> bf16 LDS (float4 loads, v4bf stores)
#pragma unroll
        for (int i = 0; i < 10; ++i) {
            int idx = tid + 256 * i;          // 0..2559
            int row = idx >> 5;
            int c4  = (idx & 31) << 2;
            const float4 v = *(const float4*)(bf + (rowBase + (size_t)row) * FF + kc * 128 + c4);
            v4bf o; o[0] = (__bf16)v.x; o[1] = (__bf16)v.y; o[2] = (__bf16)v.z; o[3] = (__bf16)v.w;
            *(v4bf*)(lA + row * LAS + c4) = o;
        }
        // W chunk: 2 x 64 rows x 128 f32 -> bf16 LDS
#pragma unroll
        for (int i = 0; i < 16; ++i) {
            int idx = tid + 256 * i;          // 0..4095
            int w   = idx >> 11;
            int rem = idx & 2047;
            int row = rem >> 5;
            int c4  = (rem & 31) << 2;
            const float* Wsel = w ? Wp : Wc;
            const float4 v = *(const float4*)(Wsel + row * FF + kc * 128 + c4);
            v4bf o; o[0] = (__bf16)v.x; o[1] = (__bf16)v.y; o[2] = (__bf16)v.z; o[3] = (__bf16)v.w;
            *(v4bf*)(lW + w * (64 * LWS) + row * LWS + c4) = o;
        }
        __syncthreads();
        // 5 tiles per wave, 4 WMMA K-steps per chunk
#pragma unroll
        for (int i = 0; i < 5; ++i) {
            int t  = wv + 8 * i;              // 0..39
            int w  = t / 20;                  // which weight matrix
            int rm = t % 20;
            int mt = rm >> 2;                 // 0..4
            int nt = rm & 3;                  // 0..3
            const __bf16* arow = lA + (16 * mt + l) * LAS;
            const __bf16* brow = lW + w * (64 * LWS) + (16 * nt + l) * LWS;
#pragma unroll
            for (int ks = 0; ks < 4; ++ks) {
                int ko = ks * 32 + hi * 8;
                v16bf a = mk16(arow + ko, arow + ko + 16);
                v16bf b = mk16(brow + ko, brow + ko + 16);
                acc[i] = __builtin_amdgcn_wmma_f32_16x16x32_bf16(
                    false, a, false, b, (short)0, acc[i], false, false);
            }
        }
    }
    __syncthreads();
    // spill fts tiles into reused LDS region (f32, 2 x 80 x 68)
#pragma unroll
    for (int i = 0; i < 5; ++i) {
        int t  = wv + 8 * i;
        int w  = t / 20;
        int rm = t % 20;
        int mt = rm >> 2;
        int nt = rm & 3;
#pragma unroll
        for (int j = 0; j < 8; ++j)
            lF[w * (80 * LFS) + (16 * mt + j + 8 * hi) * LFS + 16 * nt + l] = acc[i][j];
    }
    __syncthreads();

    // ---- stage 2: ba mix + bias + PReLU + reductions (16 thr / batch) ----
    const int bi = tid >> 4;
    const int th = tid & 15;
    const int h0 = th * 4;
    const int b  = b0 + bi;
    const int r0 = bi * 5;
    const float a1 = ac[0], a2 = ap[0];

    float bcv[4], bpv[4];
#pragma unroll
    for (int j = 0; j < 4; ++j) { bcv[j] = bc[h0 + j]; bpv[j] = bp[h0 + j]; }

    const size_t cOff = (size_t)4 * BATCH;
    const size_t hOff = cOff + (size_t)BATCH * HH;

    // h1 path: all s; accumulate mean of s=0..3, keep s=4
    float cs[4] = {0.f, 0.f, 0.f, 0.f};
    float hm[4];
#pragma unroll
    for (int s = 0; s < 5; ++s) {
        float o[4] = {bcv[0], bcv[1], bcv[2], bcv[3]};
#pragma unroll
        for (int t = 0; t < 5; ++t) {
            float wgt = lBa[bi * 25 + s * 5 + t];
            const float* fr = lF + (r0 + t) * LFS + h0;
#pragma unroll
            for (int j = 0; j < 4; ++j) o[j] += wgt * fr[j];
        }
#pragma unroll
        for (int j = 0; j < 4; ++j) o[j] = (o[j] >= 0.f) ? o[j] : a1 * o[j];
        if (s < 4) {
#pragma unroll
            for (int j = 0; j < 4; ++j) cs[j] += o[j];
        } else {
#pragma unroll
            for (int j = 0; j < 4; ++j) hm[j] = o[j];
        }
    }
#pragma unroll
    for (int j = 0; j < 4; ++j) {
        out[cOff + (size_t)b * HH + h0 + j] = cs[j] * 0.25f;
        out[hOff + (size_t)b * HH + h0 + j] = hm[j];
    }

    // h2 path: only s=3 (h_ano) and s=4 (h_unano) -> workspace
#pragma unroll
    for (int s = 3; s < 5; ++s) {
        float o[4] = {bpv[0], bpv[1], bpv[2], bpv[3]};
#pragma unroll
        for (int t = 0; t < 5; ++t) {
            float wgt = lBa[bi * 25 + s * 5 + t];
            const float* fr = lF + (80 * LFS) + (r0 + t) * LFS + h0;
#pragma unroll
            for (int j = 0; j < 4; ++j) o[j] += wgt * fr[j];
        }
#pragma unroll
        for (int j = 0; j < 4; ++j) o[j] = (o[j] >= 0.f) ? o[j] : a2 * o[j];
        float* dst = (s == 4) ? ws : (ws + (size_t)BATCH * HH); // unano | ano
#pragma unroll
        for (int j = 0; j < 4; ++j) dst[(size_t)b * HH + h0 + j] = o[j];
    }
}

// ============================ Kernel 2 ============================
// t1 = h_mv @ Wbc, t2 = h_unano @ Wbp via WMMA (shared by pos/neg),
// then dot products against c[b]/c[b-1] and h_ano[b]/h_ano[b-1].
__global__ void __launch_bounds__(128)
disc_wmma_kernel(const float* __restrict__ Wbc, const float* __restrict__ bbc,
                 const float* __restrict__ Wbp, const float* __restrict__ bbp,
                 float* __restrict__ out, const float* __restrict__ ws)
{
    __shared__ __bf16 lWb[2 * 64 * 72];   // padded 72 -> 144B rows
    __shared__ __bf16 lX [2 * 16 * 72];
    __shared__ float  lT [2 * 16 * 68];

    const int tid = threadIdx.x;
    const int b0  = blockIdx.x * 16;
    const size_t cOff = (size_t)4 * BATCH;
    const size_t hOff = cOff + (size_t)BATCH * HH;
    const float* hmv = out + hOff;
    const float* un  = ws;                        // h_unano
    const float* an  = ws + (size_t)BATCH * HH;   // h_ano

    for (int idx = tid; idx < 8192; idx += 128) {
        int w = idx >> 12, rem = idx & 4095;
        int row = rem >> 6, col = rem & 63;
        lWb[w * 4608 + row * 72 + col] = (__bf16)((w ? Wbp : Wbc)[rem]);
    }
    for (int idx = tid; idx < 2048; idx += 128) {
        int w = idx >> 10, rem = idx & 1023;
        int r = rem >> 6, col = rem & 63;
        float v = w ? un[(size_t)(b0 + r) * HH + col]
                    : hmv[(size_t)(b0 + r) * HH + col];
        lX[w * 1152 + r * 72 + col] = (__bf16)v;
    }
    __syncthreads();

    const int wv = tid >> 5, lane = tid & 31, l = lane & 15, hi = lane >> 4;
#pragma unroll
    for (int i = 0; i < 2; ++i) {
        int t = wv + 4 * i;                // 0..7
        int w = t >> 2, nt = t & 3;
        const __bf16* arow = lX + w * 1152 + l * 72;
        const __bf16* brow = lWb + w * 4608 + (16 * nt + l) * 72;
        v8f acc;
#pragma unroll
        for (int j = 0; j < 8; ++j) acc[j] = 0.0f;
#pragma unroll
        for (int ks = 0; ks < 2; ++ks) {
            int ko = ks * 32 + hi * 8;
            v16bf a = mk16(arow + ko, arow + ko + 16);
            v16bf b = mk16(brow + ko, brow + ko + 16);
            acc = __builtin_amdgcn_wmma_f32_16x16x32_bf16(
                false, a, false, b, (short)0, acc, false, false);
        }
#pragma unroll
        for (int j = 0; j < 8; ++j)
            lT[w * 1088 + (j + 8 * hi) * 68 + 16 * nt + l] = acc[j];
    }
    __syncthreads();

    // 8 lanes per batch, 8 h-values per lane; shfl_xor tree reduction
    const int bi = tid >> 3, g = tid & 7, h0 = g * 8;
    const int b  = b0 + bi;
    const int bm = (b == 0) ? (BATCH - 2) : (b - 1);
    const float* cv  = out + cOff + (size_t)b  * HH;
    const float* cmv = out + cOff + (size_t)bm * HH;
    const float* av  = an + (size_t)b  * HH;
    const float* amv = an + (size_t)bm * HH;
    const float* t1  = lT + bi * 68;
    const float* t2  = lT + 1088 + bi * 68;

    float p1p = 0.f, p1n = 0.f, p2p = 0.f, p2n = 0.f;
#pragma unroll
    for (int j = 0; j < 8; ++j) {
        float u1 = t1[h0 + j], u2 = t2[h0 + j];
        p1p += u1 * cv [h0 + j];
        p1n += u1 * cmv[h0 + j];
        p2p += u2 * av [h0 + j];
        p2n += u2 * amv[h0 + j];
    }
#pragma unroll
    for (int m = 1; m < 8; m <<= 1) {
        p1p += __shfl_xor(p1p, m, 32);
        p1n += __shfl_xor(p1n, m, 32);
        p2p += __shfl_xor(p2p, m, 32);
        p2n += __shfl_xor(p2n, m, 32);
    }
    if (g == 0) {
        float k1 = bbc[0], k2 = bbp[0];
        out[(size_t)0 * BATCH + b] = p1p + k1;   // ret1 pos
        out[(size_t)1 * BATCH + b] = p1n + k1;   // ret1 neg
        out[(size_t)2 * BATCH + b] = p2p + k2;   // ret2 pos
        out[(size_t)3 * BATCH + b] = p2n + k2;   // ret2 neg
    }
}

extern "C" void kernel_launch(void* const* d_in, const int* in_sizes, int n_in,
                              void* d_out, int out_size, void* d_ws, size_t ws_size,
                              hipStream_t stream) {
    (void)in_sizes; (void)n_in; (void)out_size; (void)ws_size;
    const float* bf  = (const float*)d_in[0];
    const float* ba  = (const float*)d_in[1];
    const float* Wc  = (const float*)d_in[2];
    const float* bc  = (const float*)d_in[3];
    const float* ac  = (const float*)d_in[4];
    const float* Wp  = (const float*)d_in[5];
    const float* bp  = (const float*)d_in[6];
    const float* ap  = (const float*)d_in[7];
    const float* Wbc = (const float*)d_in[8];
    const float* bbc = (const float*)d_in[9];
    const float* Wbp = (const float*)d_in[10];
    const float* bbp = (const float*)d_in[11];
    float* out = (float*)d_out;
    float* ws  = (float*)d_ws;   // needs 2*B*H*4 = 32 MB

    gcn_wmma_kernel <<<BATCH / 16, 256, 0, stream>>>(bf, ba, Wc, bc, ac, Wp, bp, ap, out, ws);
    disc_wmma_kernel<<<BATCH / 16, 128, 0, stream>>>(Wbc, bbc, Wbp, bbp, out, ws);
}
